// SimpleSelfAttention_28071906247236
// MI455X (gfx1250) — matrix operand order repaired and checked by reading, not compiled, not run on hardware
//
#include <hip/hip_runtime.h>

// Self-attention, B=8 S=2048 E=1024, f16 WMMA with f32 accumulation.
// Needs ws_size >= 128 MB (four 32MB f16 planes: Q, K, V^T, O).

#define DEVFN static __device__ __forceinline__

typedef __attribute__((ext_vector_type(16))) _Float16     v16h;
typedef __attribute__((ext_vector_type(8)))  float        v8f;
typedef __attribute__((ext_vector_type(4)))  unsigned int u32x4;
typedef __attribute__((ext_vector_type(2)))  unsigned int u32x2;
typedef __attribute__((ext_vector_type(4)))  float        f32x4;

constexpr int BB = 8;
constexpr int SS = 2048;
constexpr int EE = 1024;
constexpr int MT = BB * SS;   // 16384 total rows

struct Frag32 { u32x4 lo, hi; };   // 32 bytes == v16h

DEVFN unsigned short f2h(float f) {
  _Float16 h = (_Float16)f;                       // hardware RNE convert
  return __builtin_bit_cast(unsigned short, h);
}

// A-matrix 16x32 f16 layout (ISA 7.12.2): lanes 0-15 hold K {0..7,16..23},
// lanes 16-31 hold K {8..15,24..31}; M = lane%16. rowp = lane's row, 16B aligned.
DEVFN v16h load_a_frag(const unsigned short* rowp, int laneHi) {
  Frag32 t;
  t.lo = *(const u32x4*)(rowp + (laneHi ? 8 : 0));
  t.hi = *(const u32x4*)(rowp + 16 + (laneHi ? 8 : 0));
  return __builtin_bit_cast(v16h, t);
}
// B-matrix 32x16 f16 layout: lane n holds column n over K; lanes 0-15 K {0..7,16..23},
// lanes 16-31 K {8..15,24..31}. rowp points at the source row that forms column n.
DEVFN v16h load_b_frag(const unsigned short* rowp, int laneHi) {
  Frag32 t;
  t.lo = *(const u32x4*)(rowp + (laneHi ? 8 : 0));
  t.hi = *(const u32x4*)(rowp + 16 + (laneHi ? 8 : 0));
  return __builtin_bit_cast(v16h, t);
}
DEVFN v8f wmma16(v16h a, v16h b, v8f c) {
  return __builtin_amdgcn_wmma_f32_16x16x32_f16(false, a, false, b,
                                                (short)0, c, false, false);
}

// ---------------------------------------------------------------------------
// Y[m,n] = sum_k X[m,k] * W[n,k]   (torch Linear: x @ W^T), K = EE = 1024.
// OMODE 0: Y f16 [M,E];  1: Y f16 transposed per-batch [b][n][s];  2: f32 + bias.
// Block: 256 thr = 8 waves; C tile 128x128; each wave 64x32 (4x2 WMMA frags).
// ---------------------------------------------------------------------------
template<bool XHALF, int OMODE>
__global__ void __launch_bounds__(256)
gemm_xwT(const void* __restrict__ Xv, const float* __restrict__ W,
         void* __restrict__ Yv, const float* __restrict__ bias) {
  constexpr int LDT = 40;                       // 32 + 8 halves pad (80B rows)
  __shared__ __align__(16) unsigned short Xs[128 * LDT];
  __shared__ __align__(16) unsigned short Ws[128 * LDT];
  const int tid    = threadIdx.x;
  const int lane   = tid & 31;
  const int w      = tid >> 5;
  const int laneHi = lane >> 4;
  const int ln     = lane & 15;
  const int m0 = blockIdx.y * 128;
  const int n0 = blockIdx.x * 128;
  const int wm = (w & 1) * 64;
  const int wn = (w >> 1) * 32;
  v8f acc[4][2] = {};
  const float*          Xf = (const float*)Xv;
  const unsigned short* Xh = (const unsigned short*)Xv;

  for (int k0 = 0; k0 < EE; k0 += 32) {
    // Stage 128x32 X and W tiles as f16 (each thread: 4 quads of each).
#pragma unroll
    for (int i = 0; i < 4; ++i) {
      const int idx4 = tid + i * 256;           // 0..1023
      const int r    = idx4 >> 3;               // row 0..127
      const int c4   = (idx4 & 7) * 4;          // col 0,4,..,28
      if (XHALF) {
        u32x2 d = *(const u32x2*)(Xh + (size_t)(m0 + r) * EE + k0 + c4);
        *(u32x2*)&Xs[r * LDT + c4] = d;
      } else {
        f32x4 d = *(const f32x4*)(Xf + (size_t)(m0 + r) * EE + k0 + c4);
        u32x2 p = { (unsigned)f2h(d[0]) | ((unsigned)f2h(d[1]) << 16),
                    (unsigned)f2h(d[2]) | ((unsigned)f2h(d[3]) << 16) };
        *(u32x2*)&Xs[r * LDT + c4] = p;
      }
      f32x4 dw = *(const f32x4*)(W + (size_t)(n0 + r) * EE + k0 + c4);
      u32x2 pw = { (unsigned)f2h(dw[0]) | ((unsigned)f2h(dw[1]) << 16),
                   (unsigned)f2h(dw[2]) | ((unsigned)f2h(dw[3]) << 16) };
      *(u32x2*)&Ws[r * LDT + c4] = pw;
    }
    // Prefetch next K-chunk (one 64/128B row-chunk per thread, global_prefetch).
    if (k0 + 32 < EE) {
      if (tid < 128) {
        if (XHALF) __builtin_prefetch(Xh + (size_t)(m0 + tid) * EE + k0 + 32, 0, 1);
        else       __builtin_prefetch(Xf + (size_t)(m0 + tid) * EE + k0 + 32, 0, 1);
      } else {
        __builtin_prefetch(W + (size_t)(n0 + tid - 128) * EE + k0 + 32, 0, 1);
      }
    }
    __syncthreads();
    v16h a[4], b[2];
#pragma unroll
    for (int mt = 0; mt < 4; ++mt)
      a[mt] = load_a_frag(&Xs[(wm + mt * 16 + ln) * LDT], laneHi);
#pragma unroll
    for (int nt = 0; nt < 2; ++nt)
      b[nt] = load_b_frag(&Ws[(wn + nt * 16 + ln) * LDT], laneHi);
#pragma unroll
    for (int mt = 0; mt < 4; ++mt)
#pragma unroll
      for (int nt = 0; nt < 2; ++nt)
        acc[mt][nt] = wmma16(a[mt], b[nt], acc[mt][nt]);
    __syncthreads();
  }

  // Epilogue. C layout: VGPR j = row j (lanes 0-15) / j+8 (lanes 16-31), col = lane%16.
#pragma unroll
  for (int mt = 0; mt < 4; ++mt) {
#pragma unroll
    for (int nt = 0; nt < 2; ++nt) {
      const int ncol  = n0 + wn + nt * 16 + ln;
      const int mbase = m0 + wm + mt * 16 + laneHi * 8;
#pragma unroll
      for (int j = 0; j < 8; ++j) {
        const int   m = mbase + j;
        const float v = acc[mt][nt][j];
        if constexpr (OMODE == 0) {
          ((unsigned short*)Yv)[(size_t)m * EE + ncol] = f2h(v);
        } else if constexpr (OMODE == 1) {
          const int bb = m >> 11, ss = m & (SS - 1);
          ((unsigned short*)Yv)[((size_t)bb * EE + ncol) * SS + ss] = f2h(v);
        } else {
          ((float*)Yv)[(size_t)m * EE + ncol] = v + bias[ncol];
        }
      }
    }
  }
}

// ---------------------------------------------------------------------------
// Fused flash attention: one block per (batch, 16 query rows). 8 waves; each
// wave owns a 128-wide slice of the 1024 head dim for V/O; QK^T partials are
// reduced deterministically via LDS. Online softmax over 32-key steps.
// ---------------------------------------------------------------------------
__global__ void __launch_bounds__(256)
flash_attn(const unsigned short* __restrict__ Q,
           const unsigned short* __restrict__ Kp,
           const unsigned short* __restrict__ VT,   // [b][e][s]
           unsigned short* __restrict__ O) {
  constexpr int   LDQ   = 1032;                 // 1024 + 8 halves pad
  constexpr int   LDV   = 40;
  constexpr int   LDP   = 40;
  constexpr float SCALE = 0.03125f;             // 1/sqrt(1024)
  __shared__ __align__(16) unsigned short Qs[16 * LDQ];     // 33.0 KB
  __shared__ __align__(16) unsigned short KV[40960];        // 80 KB: K 32xLDQ or V^T 1024xLDV
  __shared__ __align__(16) unsigned short P[16 * LDP];      // probs tile
  __shared__ float Spart[8][16][32];                        // per-wave QK^T partials
  __shared__ float Ssc[16][33];
  __shared__ float mstat[16], lstat[16], alphas[16];

  const int tid    = threadIdx.x;
  const int lane   = tid & 31;
  const int w      = tid >> 5;
  const int laneHi = lane >> 4;
  const int ln     = lane & 15;
  const int b  = blockIdx.x >> 7;
  const int q0 = (blockIdx.x & 127) * 16;
  const unsigned short* Qb = Q  + ((size_t)b * SS + q0) * EE;
  const unsigned short* Kb = Kp + (size_t)b * SS * EE;
  const unsigned short* Vb = VT + (size_t)b * EE * SS;

  // Stage Q tile 16x1024 (2048 b128 loads / 256 threads).
#pragma unroll
  for (int i = 0; i < 8; ++i) {
    const int idx = tid + i * 256;
    const int r = idx >> 7, c = (idx & 127) * 8;
    *(u32x4*)&Qs[r * LDQ + c] = *(const u32x4*)(Qb + (size_t)r * EE + c);
  }
  if (tid < 16) { mstat[tid] = -1e30f; lstat[tid] = 0.f; }

  v8f acc[8] = {};                              // wave's 16 x 128 O slice
  const int dws = w * 128;

  for (int kt = 0; kt < SS / 32; ++kt) {
    __syncthreads();  // prev iteration done with KV; Qs/stats visible
    // Stage K tile 32x1024.
    const unsigned short* Ksrc = Kb + (size_t)(kt * 32) * EE;
#pragma unroll
    for (int i = 0; i < 16; ++i) {
      const int idx = tid + i * 256;
      const int r = idx >> 7, c = (idx & 127) * 8;
      *(u32x4*)&KV[r * LDQ + c] = *(const u32x4*)(Ksrc + (size_t)r * EE + c);
    }
    // Prefetch next step's K (contiguous, 128B granules) and V^T (one 64B row
    // chunk per prefetch; rows are strided 4KB) while this step computes.
    if (kt + 1 < SS / 32) {
      const unsigned short* Kn = Kb + (size_t)((kt + 1) * 32) * EE;
#pragma unroll
      for (int i = 0; i < 2; ++i)
        __builtin_prefetch(Kn + (size_t)(tid + i * 256) * 64, 0, 1);
      const unsigned short* Vn = Vb + (size_t)((kt + 1) * 32);
#pragma unroll
      for (int i = 0; i < 4; ++i)
        __builtin_prefetch(Vn + (size_t)(tid + i * 256) * SS, 0, 1);
    }
    __syncthreads();
    // Partial S = Q[:, dws:dws+128] * K^T  (two 16-col halves, 4 K-chunks).
    v8f sa0 = {}, sa1 = {};
#pragma unroll
    for (int kc = 0; kc < 4; ++kc) {
      const int koff = dws + kc * 32;
      v16h aq = load_a_frag(&Qs[ln * LDQ + koff], laneHi);
      v16h b0 = load_b_frag(&KV[ln * LDQ + koff], laneHi);
      v16h b1 = load_b_frag(&KV[(16 + ln) * LDQ + koff], laneHi);
      sa0 = wmma16(aq, b0, sa0);
      sa1 = wmma16(aq, b1, sa1);
    }
#pragma unroll
    for (int j = 0; j < 8; ++j) {
      Spart[w][j + laneHi * 8][ln]      = sa0[j];
      Spart[w][j + laneHi * 8][16 + ln] = sa1[j];
    }
    __syncthreads();
    // Deterministic cross-wave reduction of partials; stage V^T tile (KV free).
    for (int idx = tid; idx < 512; idx += 256) {
      const int r = idx >> 5, c = idx & 31;
      float t = 0.f;
#pragma unroll
      for (int ww = 0; ww < 8; ++ww) t += Spart[ww][r][c];
      Ssc[r][c] = t;
    }
    const unsigned short* Vsrc = Vb + (size_t)(kt * 32);
#pragma unroll
    for (int i = 0; i < 16; ++i) {
      const int idx = tid + i * 256;
      const int r = idx >> 2, c = (idx & 3) * 8;
      *(u32x4*)&KV[r * LDV + c] = *(const u32x4*)(Vsrc + (size_t)r * SS + c);
    }
    __syncthreads();
    // Online softmax on wave 0: row = lane%16, each lane-half scans 16 cols,
    // combine max/sum across halves with a xor-16 shuffle.
    if (w == 0) {
      const int r = ln;
      const int cb = laneHi * 16;
      const float mo = mstat[r];
      float mx = mo;
#pragma unroll
      for (int j = 0; j < 16; ++j) mx = fmaxf(mx, Ssc[r][cb + j] * SCALE);
      mx = fmaxf(mx, __shfl_xor(mx, 16, 32));
      float ls = 0.f;
#pragma unroll
      for (int j = 0; j < 16; ++j) {
        const float p = __expf(Ssc[r][cb + j] * SCALE - mx);
        ls += p;
        P[r * LDP + cb + j] = f2h(p);
      }
      ls += __shfl_xor(ls, 16, 32);
      if (laneHi == 0) {
        const float al = __expf(mo - mx);
        mstat[r]  = mx;
        lstat[r]  = al * lstat[r] + ls;
        alphas[r] = al;
      }
    }
    __syncthreads();
    // Rescale accumulators by alpha(row) and add P @ V for this wave's d-slice.
    float av[8];
#pragma unroll
    for (int j = 0; j < 8; ++j) av[j] = alphas[j + laneHi * 8];
    v16h ap = load_a_frag(&P[ln * LDP], laneHi);
#pragma unroll
    for (int nt = 0; nt < 8; ++nt) {
#pragma unroll
      for (int j = 0; j < 8; ++j) acc[nt][j] *= av[j];
      v16h bv = load_b_frag(&KV[(dws + nt * 16 + ln) * LDV], laneHi);
      acc[nt] = wmma16(ap, bv, acc[nt]);
    }
  }

  // Normalize by l and write O (f16 [b][s][e]).
  float lv[8];
#pragma unroll
  for (int j = 0; j < 8; ++j) lv[j] = 1.0f / lstat[j + laneHi * 8];
  unsigned short* Ob = O + ((size_t)b * SS + q0) * EE;
#pragma unroll
  for (int nt = 0; nt < 8; ++nt) {
    const int col = dws + nt * 16 + ln;
#pragma unroll
    for (int j = 0; j < 8; ++j)
      Ob[(size_t)(j + laneHi * 8) * EE + col] = f2h(acc[nt][j] * lv[j]);
  }
}

extern "C" void kernel_launch(void* const* d_in, const int* in_sizes, int n_in,
                              void* d_out, int out_size, void* d_ws, size_t ws_size,
                              hipStream_t stream) {
  (void)in_sizes; (void)n_in; (void)out_size; (void)ws_size;
  const float* value = (const float*)d_in[0];
  const float* key_  = (const float*)d_in[1];
  const float* query = (const float*)d_in[2];
  const float* Wq    = (const float*)d_in[3];
  const float* Wk    = (const float*)d_in[4];
  const float* Wv    = (const float*)d_in[5];
  const float* Wo    = (const float*)d_in[6];
  const float* bo    = (const float*)d_in[7];
  float* out = (float*)d_out;

  const size_t plane = (size_t)MT * EE;         // 16M f16 elems = 32 MB
  unsigned short* wsQ  = (unsigned short*)d_ws;
  unsigned short* wsK  = wsQ + plane;
  unsigned short* wsVT = wsK + plane;
  unsigned short* wsO  = wsVT + plane;

  dim3 gG(EE / 128, MT / 128);                  // (8, 128)
  gemm_xwT<false, 0><<<gG, 256, 0, stream>>>(query, Wq, wsQ,  nullptr);
  gemm_xwT<false, 0><<<gG, 256, 0, stream>>>(key_,  Wk, wsK,  nullptr);
  gemm_xwT<false, 1><<<gG, 256, 0, stream>>>(value, Wv, wsVT, nullptr);   // V^T
  flash_attn<<<dim3(BB * (SS / 16)), 256, 0, stream>>>(wsQ, wsK, wsVT, wsO);
  gemm_xwT<true, 2><<<gG, 256, 0, stream>>>(wsO, Wo, out, bo);
}
